// hypercorre_topk2_82008105550138
// MI455X (gfx1250) — matrix operand order repaired and checked by reading, not compiled, not run on hardware
//
#include <hip/hip_runtime.h>

typedef __bf16 bf16_t;
typedef __attribute__((ext_vector_type(16))) __bf16 v16bf;
typedef __attribute__((ext_vector_type(8)))  float  v8f;

#define C_DIM 256
#define NQ    4096
#define NR    4096
#define T_DIM 3
#define B_DIM 2
#define SIMSTRIDE 36   // bf16 elements per LDS sim column (32 rows + pad, 72B: 8B-aligned, bank-spread)

union FragU { v16bf v; uint4 u[2]; };

// A-matrix 16x32 bf16 fragment (ISA 7.12.2): lanes 0-15 hold M=lane, K={0..7,16..23};
// lanes 16-31 hold M=lane-16, K={8..15,24..31}. Row-major source, stride ld elements.
__device__ __forceinline__ v16bf load_fragA_bf16(const bf16_t* __restrict__ src, int lane, int ld) {
  const int m  = lane & 15;
  const int kb = (lane & 16) ? 8 : 0;
  const bf16_t* p = src + (size_t)m * ld + kb;
  FragU f;
  f.u[0] = *(const uint4*)(p);        // K = kb .. kb+7
  f.u[1] = *(const uint4*)(p + 16);   // K = kb+16 .. kb+23
  return f.v;
}

// Same A layout but built from an f32 row-major source (convert on load).
__device__ __forceinline__ v16bf load_fragA_f32(const float* __restrict__ src, int lane, int ld) {
  const int m  = lane & 15;
  const int kb = (lane & 16) ? 8 : 0;
  const float* p = src + (size_t)m * ld + kb;
  v16bf v;
#pragma unroll
  for (int i = 0; i < 8; ++i) v[i] = (__bf16)p[i];
#pragma unroll
  for (int i = 0; i < 8; ++i) v[8 + i] = (__bf16)p[16 + i];
  return v;
}

// B-matrix 32x16 bf16 fragment: lane n (n<16) holds K={0..7,16..23} of column n,
// lane n+16 holds K={8..15,24..31} of column n (same K order as the A fragment).
// Source is the TRANSPOSED matrix stored row-major as [N, K].
__device__ __forceinline__ v16bf load_fragB_bf16(const bf16_t* __restrict__ srcT, int lane, int ld) {
  const int n  = lane & 15;
  const int kb = (lane & 16) ? 8 : 0;
  const bf16_t* p = srcT + (size_t)n * ld + kb;
  FragU f;
  f.u[0] = *(const uint4*)(p);       // K = kb .. kb+7
  f.u[1] = *(const uint4*)(p + 16);  // K = kb+16 .. kb+23
  return f.v;
}

__device__ __forceinline__ unsigned pack_bf16x2(float a, float b) {
  union { __bf16 h[2]; unsigned u; } t;
  t.h[0] = (__bf16)a; t.h[1] = (__bf16)b;
  return t.u;
}

// ---------------- kernel 1: W [C,C] f32 -> W^T [C,C] bf16 ----------------
__global__ void wt_bf16_kernel(const float* __restrict__ W, bf16_t* __restrict__ WT) {
  const int j = blockIdx.x, c = threadIdx.x;
  WT[(size_t)j * C_DIM + c] = (__bf16)W[(size_t)c * C_DIM + j];
}

// ------------- kernel 2: P = l2norm(X @ W + b), emitted as bf16 -------------
// 256 threads = 8 waves; each wave owns 16 rows; workgroup = 128 rows.
__global__ __launch_bounds__(256) void proj_l2norm_kernel(
    const float* __restrict__ X, const bf16_t* __restrict__ WT,
    const float* __restrict__ bias, bf16_t* __restrict__ out) {
  extern __shared__ char smem[];
  float* rowbuf  = (float*)smem;            // 128 * 256 f32
  float* partial = rowbuf + 128 * C_DIM;    // 256 f32

  const int tid  = threadIdx.x;
  const int wave = tid >> 5, lane = tid & 31;
  const int rowbase = blockIdx.x * 128;
  const int wrow = rowbase + wave * 16;

  v16bf a[8];
#pragma unroll
  for (int kc = 0; kc < 8; ++kc)
    a[kc] = load_fragA_f32(X + (size_t)wrow * C_DIM + kc * 32, lane, C_DIM);

  const int n  = lane & 15;
  const int mb = (lane & 16) ? 8 : 0;
  for (int jt = 0; jt < 16; ++jt) {
    v8f c = {};
#pragma unroll
    for (int kc = 0; kc < 8; ++kc) {
      v16bf fb = load_fragB_bf16(WT + (size_t)(jt * 16) * C_DIM + kc * 32, lane, C_DIM);
      c = __builtin_amdgcn_wmma_f32_16x16x32_bf16(false, a[kc], false, fb, (short)0, c, false, false);
    }
    const float bb = bias[jt * 16 + n];
#pragma unroll
    for (int v = 0; v < 8; ++v)
      rowbuf[(wave * 16 + v + mb) * C_DIM + jt * 16 + n] = c[v] + bb;
  }
  __syncthreads();

  const int r = tid >> 1, h = tid & 1;
  const float* rb = rowbuf + r * C_DIM + h * 128;
  float s = 0.f;
  for (int i = 0; i < 128; ++i) { const float x = rb[i]; s += x * x; }
  partial[tid] = s;
  __syncthreads();
  const float scale = rsqrtf(partial[2 * r] + partial[2 * r + 1] + 1e-12f);
  bf16_t* op = out + (size_t)(rowbase + r) * C_DIM + h * 128;
  for (int i = 0; i < 128; ++i) op[i] = (__bf16)(rb[i] * scale);
}

// ------ kernel 3: fused sim GEMM + top-16 + softmax + gather-aggregate ------
// Workgroup = one (b, t, 32-query tile). Each B fragment feeds TWO accumulator
// tiles (2x arithmetic intensity on K traffic). sim kept as bf16 in LDS,
// column-major (stride 36) so 8 consecutive rows/lane pack into b64 stores.
__global__ __launch_bounds__(256) void hypercorr_fused_kernel(
    const bf16_t* __restrict__ qb, const bf16_t* __restrict__ kb,
    const float* __restrict__ ref, float* __restrict__ out) {
  extern __shared__ char smem[];
  bf16_t*   simb = (bf16_t*)smem;                       // NR * SIMSTRIDE bf16 = 288 KB
  unsigned* mrg  = (unsigned*)(smem + NR * SIMSTRIDE * 2); // 32*128 packed (idx<<16|bf16) = 16 KB
  float*    fw   = (float*)(mrg + 32 * 128);            // 32*16 softmax weights
  int*      fi   = (int*)(fw + 32 * 16);                // 32*16 final indices

  const int tid  = threadIdx.x;
  const int wave = tid >> 5, lane = tid & 31;
  const int q0 = blockIdx.x * 32;
  const int t  = blockIdx.y, b = blockIdx.z;
  const bf16_t* qptr   = qb  + ((size_t)b * NQ + q0) * C_DIM;
  const bf16_t* kptr   = kb  + ((size_t)(b * T_DIM + t) * NR) * C_DIM;
  const float*  refptr = ref + ((size_t)(b * T_DIM + t) * NR) * C_DIM;

  // Resident A fragments for two 16-row tiles (shared by all 8 waves).
  v16bf a0[8], a1[8];
#pragma unroll
  for (int kc = 0; kc < 8; ++kc) {
    a0[kc] = load_fragA_bf16(qptr + kc * 32, lane, C_DIM);
    a1[kc] = load_fragA_bf16(qptr + 16 * C_DIM + kc * 32, lane, C_DIM);
  }

  const int n  = lane & 15;
  const int mb = (lane & 16) ? 8 : 0;
  // Phase 1: each wave sweeps 32 column tiles of 16 (all 4096 ref tokens covered).
  for (int it = 0; it < 32; ++it) {
    const int r0 = (it * 8 + wave) * 16;
    v8f c0 = {}, c1 = {};
#pragma unroll
    for (int kc = 0; kc < 8; ++kc) {
      v16bf fb = load_fragB_bf16(kptr + (size_t)r0 * C_DIM + kc * 32, lane, C_DIM);
      c0 = __builtin_amdgcn_wmma_f32_16x16x32_bf16(false, a0[kc], false, fb, (short)0, c0, false, false);
      c1 = __builtin_amdgcn_wmma_f32_16x16x32_bf16(false, a1[kc], false, fb, (short)0, c1, false, false);
    }
    // Column-major bf16 store: lane holds rows mb..mb+7 (c0) and 16+mb..16+mb+7 (c1)
    // of column r0+n -> two aligned b64 stores per accumulator.
    bf16_t* colp = simb + (size_t)(r0 + n) * SIMSTRIDE;
    uint2 p0 = { pack_bf16x2(c0[0], c0[1]), pack_bf16x2(c0[2], c0[3]) };
    uint2 p1 = { pack_bf16x2(c0[4], c0[5]), pack_bf16x2(c0[6], c0[7]) };
    uint2 p2 = { pack_bf16x2(c1[0], c1[1]), pack_bf16x2(c1[2], c1[3]) };
    uint2 p3 = { pack_bf16x2(c1[4], c1[5]), pack_bf16x2(c1[6], c1[7]) };
    *(uint2*)(colp + mb)          = p0;
    *(uint2*)(colp + mb + 4)      = p1;
    *(uint2*)(colp + 16 + mb)     = p2;
    *(uint2*)(colp + 16 + mb + 4) = p3;
  }
  __syncthreads();

  // Phase 2: segmented top-16 — 8 threads per row, stride-8 interleaved columns.
  {
    const int row = tid >> 3, seg = tid & 7;
    float bv[16]; int bi[16];
#pragma unroll
    for (int s = 0; s < 16; ++s) { bv[s] = -3.0e38f; bi[s] = 0; }
    for (int i = 0; i < 512; ++i) {
      const int col = seg + (i << 3);
      const float v = (float)simb[(size_t)col * SIMSTRIDE + row];
      if (v > bv[15]) {               // running 16th-largest threshold
        float cv = v; int ci = col;
#pragma unroll
        for (int s = 0; s < 16; ++s) {       // sorted insertion, no dyn indexing
          const bool sw = cv > bv[s];
          const float tv = sw ? bv[s] : cv; const int ti = sw ? bi[s] : ci;
          bv[s] = sw ? cv : bv[s];          bi[s] = sw ? ci : bi[s];
          cv = tv; ci = ti;
        }
      }
    }
#pragma unroll
    for (int s = 0; s < 16; ++s) {
      unsigned hb = (unsigned)__builtin_bit_cast(unsigned short, (__bf16)bv[s]);
      mrg[row * 128 + seg * 16 + s] = ((unsigned)bi[s] << 16) | hb;
    }
  }
  __syncthreads();

  // Phase 3: per-row merge of 8 partial lists + softmax.
  if (tid < 32) {
    const int row = tid;
    float bv[16]; int bi[16];
#pragma unroll
    for (int s = 0; s < 16; ++s) { bv[s] = -3.0e38f; bi[s] = 0; }
    for (int i = 0; i < 128; ++i) {
      const unsigned pk = mrg[row * 128 + i];
      const float v = (float)__builtin_bit_cast(__bf16, (unsigned short)(pk & 0xffffu));
      if (v > bv[15]) {
        float cv = v; int ci = (int)(pk >> 16);
#pragma unroll
        for (int s = 0; s < 16; ++s) {
          const bool sw = cv > bv[s];
          const float tv = sw ? bv[s] : cv; const int ti = sw ? bi[s] : ci;
          bv[s] = sw ? cv : bv[s];          bi[s] = sw ? ci : bi[s];
          cv = tv; ci = ti;
        }
      }
    }
    const float mx = bv[0];            // list is sorted descending
    float e[16]; float sum = 0.f;
#pragma unroll
    for (int s = 0; s < 16; ++s) { e[s] = __expf(bv[s] - mx); sum += e[s]; }
    const float inv = 1.0f / sum;
#pragma unroll
    for (int s = 0; s < 16; ++s) { fw[row * 16 + s] = e[s] * inv; fi[row * 16 + s] = bi[s]; }
  }
  __syncthreads();

  // Phase 4: gather matched ref rows + weighted aggregation (f32 output).
  {
    const int row = tid >> 3, cb = tid & 7;   // 8 threads/row, 32 channels each
    float acc[32];
#pragma unroll
    for (int i = 0; i < 32; ++i) acc[i] = 0.f;
#pragma unroll
    for (int s = 0; s < 16; ++s) {
      const float w = fw[row * 16 + s];
      const float* rp = refptr + (size_t)fi[row * 16 + s] * C_DIM + cb * 32;
#pragma unroll
      for (int i = 0; i < 32; ++i) acc[i] += w * rp[i];
    }
    float* op = out + (((size_t)(b * T_DIM + t) * NQ) + q0 + row) * C_DIM + cb * 32;
#pragma unroll
    for (int i = 0; i < 32; ++i) op[i] = acc[i];
  }
}

extern "C" void kernel_launch(void* const* d_in, const int* in_sizes, int n_in,
                              void* d_out, int out_size, void* d_ws, size_t ws_size,
                              hipStream_t stream) {
  (void)in_sizes; (void)n_in; (void)out_size; (void)ws_size;
  const float* tgt = (const float*)d_in[0];   // [B, Nq, C]
  const float* ref = (const float*)d_in[1];   // [B, T, Nr, C]
  const float* Wq  = (const float*)d_in[2];   // [C, C]
  const float* bq  = (const float*)d_in[3];   // [C]
  const float* Wk  = (const float*)d_in[4];   // [C, C]
  const float* bk  = (const float*)d_in[5];   // [C]
  float* out = (float*)d_out;                 // [B, T, Nq, C] f32

  // Workspace layout (bytes): WqT 128K | WkT 128K | q_bf16 4M | k_bf16 12M
  char* ws = (char*)d_ws;
  bf16_t* wqT  = (bf16_t*)(ws);
  bf16_t* wkT  = (bf16_t*)(ws + 131072);
  bf16_t* qbuf = (bf16_t*)(ws + 262144);
  bf16_t* kbuf = (bf16_t*)(ws + 262144 + 4194304);

  wt_bf16_kernel<<<256, 256, 0, stream>>>(Wq, wqT);
  wt_bf16_kernel<<<256, 256, 0, stream>>>(Wk, wkT);

  const size_t projShm = (size_t)(128 * C_DIM + 256) * sizeof(float);
  proj_l2norm_kernel<<<64, 256, projShm, stream>>>(tgt, wqT, bq, qbuf);     // 8192 rows
  proj_l2norm_kernel<<<192, 256, projShm, stream>>>(ref, wkT, bk, kbuf);    // 24576 rows

  const size_t fusedShm = (size_t)NR * SIMSTRIDE * 2        // bf16 sim
                        + (size_t)32 * 128 * 4              // packed merge lists
                        + (size_t)32 * 16 * 4               // fw
                        + (size_t)32 * 16 * 4;              // fi
  hypercorr_fused_kernel<<<dim3(NQ / 32, T_DIM, B_DIM), 256, fusedShm, stream>>>(qbuf, kbuf, ref, out);
}
